// AttentionLSTM_13786845020300
// MI455X (gfx1250) — compile-verified, matrix operand order checked by reading, not run on hardware
//
#include <hip/hip_runtime.h>
#include <math.h>

#define NB   128   // persistent grid: 128 blocks (co-resident on MI455X)
#define TPB  256   // 8 waves of 32
#define N_   128
#define T_   512
#define D_   512
#define H_   512
#define FH   2048  // 4*H

typedef __attribute__((ext_vector_type(16))) __bf16 bf16x16;
typedef __attribute__((ext_vector_type(8)))  __bf16 bf16x8;
typedef __attribute__((ext_vector_type(8)))  float  f32x8;

// round-to-nearest-even fp32 -> bf16
__device__ inline unsigned short f2bf(float f) {
  union { float f; unsigned u; } v; v.f = f;
  unsigned r = v.u + 0x7FFFu + ((v.u >> 16) & 1u);
  return (unsigned short)(r >> 16);
}

// 16-bit A/B fragment (16x32), bf16 row-major source, ld in elements.
// Lane layout per CDNA5 ISA: lanes 0-15 -> K {0..7,16..23}, lanes 16-31 -> K {8..15,24..31}.
__device__ inline bf16x16 frag_bf(const unsigned short* base, int row, int ld, int k0, int lane) {
  int koff = (lane & 16) ? 8 : 0;
  const bf16x8* p = (const bf16x8*)(base + (size_t)row * ld + k0 + koff);
  union { bf16x16 v; bf16x8 h[2]; } u;
  u.h[0] = p[0];   // K = koff .. koff+7
  u.h[1] = p[2];   // K = 16+koff .. 16+koff+7
  return u.v;
}

// device-wide split barrier via L2 atomics (all NB blocks are co-resident)
__device__ inline void grid_sync(unsigned* cnt, unsigned* gen) {
  __syncthreads();
  if (threadIdx.x == 0) {
    __threadfence();
    unsigned g = __hip_atomic_load(gen, __ATOMIC_ACQUIRE, __HIP_MEMORY_SCOPE_AGENT);
    unsigned a = __hip_atomic_fetch_add(cnt, 1u, __ATOMIC_ACQ_REL, __HIP_MEMORY_SCOPE_AGENT);
    if (a == NB - 1) {
      __hip_atomic_store(cnt, 0u, __ATOMIC_RELAXED, __HIP_MEMORY_SCOPE_AGENT);
      __hip_atomic_fetch_add(gen, 1u, __ATOMIC_ACQ_REL, __HIP_MEMORY_SCOPE_AGENT);
    } else {
      while (__hip_atomic_load(gen, __ATOMIC_ACQUIRE, __HIP_MEMORY_SCOPE_AGENT) == g) {
        __builtin_amdgcn_s_sleep(2);
      }
    }
  }
  __syncthreads();
}

__global__ void __launch_bounds__(TPB, 1)
attn_lstm_persistent(const float* __restrict__ x, const float* __restrict__ A,
                     const float* __restrict__ Wx, const float* __restrict__ Wh,
                     const float* __restrict__ Wattn, const float* __restrict__ b,
                     float* __restrict__ out,
                     unsigned* bar, float* h_ws, float* c_ws, float* a_ws,
                     unsigned short* hbf, unsigned short* attnbf, unsigned short* xbf,
                     unsigned short* WxT, unsigned short* WhT, unsigned short* WattnT) {
  const int tid  = threadIdx.x;
  const int gtid = blockIdx.x * TPB + tid;           // 0..32767
  unsigned* cnt = bar;
  unsigned* gen = bar + 1;

  __shared__ float sred[TPB * 16];                   // 16 KB attention reduction
  __shared__ float wsm[16];

  // ---------- init: transpose weights to bf16 (4H x K), h0 = c0 = mean(Aflat) ----------
  for (int i = gtid; i < FH * H_; i += NB * TPB) {
    int nout = i >> 9;                               // output column
    int k    = i & 511;                              // K index (contiguous in WT)
    int src  = k * FH + nout;
    WxT[i]    = f2bf(Wx[src]);
    WhT[i]    = f2bf(Wh[src]);
    WattnT[i] = f2bf(Wattn[src]);
  }
  for (int i = gtid; i < N_ * H_; i += NB * TPB) {
    const float* ap = A + (size_t)i * 16;
    float s = 0.f;
#pragma unroll
    for (int k = 0; k < 16; ++k) s += ap[k];
    s *= (1.0f / 16.0f);
    h_ws[i] = s;
    c_ws[i] = s;
  }
  grid_sync(cnt, gen);

  // GEMM tile assignment: 8 row-tiles x 16 col-groups = 128 blocks, 8 waves = 8 col tiles
  const int lane  = tid & 31;
  const int wave  = tid >> 5;
  const int rt    = blockIdx.x >> 4;                 // 0..7   (rows of N)
  const int cg    = blockIdx.x & 15;                 // 0..15  (column group)
  const int rbase = rt * 16;
  const int cbase = (cg * 8 + wave) * 16;            // 0..2032
  const int arow  = rbase + (lane & 15);
  const int brow  = cbase + (lane & 15);
  const int srow  = (lane & 16) ? 8 : 0;             // C/D row offset per lane half
  const int scol  = lane & 15;

  const float scale = 0.044194173824159216f;         // 1/sqrt(512)

  for (int t = 0; t < T_; ++t) {
    // -------- phase A: stage xt as bf16 (once, not per column-group); 16-slot attention --------
    {
      // xt -> xbf: 65536 elements over 32768 threads, contiguous within each row
#pragma unroll
      for (int e = 0; e < 2; ++e) {
        int idx = gtid + e * (NB * TPB);             // 0..65535 -> (n, d)
        int nn = idx >> 9;
        int d  = idx & 511;
        xbf[idx] = f2bf(x[(size_t)nn * T_ * D_ + (size_t)t * D_ + d]);
      }

      const int n = blockIdx.x;
      const float hv0 = h_ws[n * H_ + tid];
      const float hv1 = h_ws[n * H_ + tid + 256];
      hbf[n * H_ + tid]       = f2bf(hv0);
      hbf[n * H_ + tid + 256] = f2bf(hv1);
      const float* a0 = A + ((size_t)(n * H_ + tid)) * 16;
      const float* a1 = A + ((size_t)(n * H_ + tid + 256)) * 16;
#pragma unroll
      for (int k = 0; k < 16; ++k)
        sred[tid * 16 + k] = hv0 * a0[k] + hv1 * a1[k];
      __syncthreads();
      for (int off = 128; off > 0; off >>= 1) {
        if (tid < off) {
#pragma unroll
          for (int k = 0; k < 16; ++k)
            sred[tid * 16 + k] += sred[(tid + off) * 16 + k];
        }
        __syncthreads();
      }
      if (tid == 0) {
        float mx = -1e30f;
        for (int k = 0; k < 16; ++k) mx = fmaxf(mx, sred[k] * scale);
        float e[16]; float ssum = 0.f;
        for (int k = 0; k < 16; ++k) { e[k] = __expf(sred[k] * scale - mx); ssum += e[k]; }
        float inv = 1.0f / ssum;
        for (int k = 0; k < 16; ++k) wsm[k] = e[k] * inv;
      }
      __syncthreads();
      float at0 = 0.f, at1 = 0.f;
#pragma unroll
      for (int k = 0; k < 16; ++k) { at0 += a0[k] * wsm[k]; at1 += a1[k] * wsm[k]; }
      attnbf[n * H_ + tid]       = f2bf(at0);
      attnbf[n * H_ + tid + 256] = f2bf(at1);
    }
    grid_sync(cnt, gen);

    // -------- phase B: a = xt@Wx + h@Wh + attn@Wattn, 3 independent WMMA chains --------
    {
      f32x8 accx = {};
      f32x8 acch = {};
      f32x8 acca = {};
#pragma unroll 2
      for (int k0 = 0; k0 < 512; k0 += 32) {
        if (k0 + 32 < 512) {                         // prefetch next K-slice of weights (L2)
          __builtin_prefetch(WxT + (size_t)brow * H_ + k0 + 32, 0, 1);
          __builtin_prefetch(WhT + (size_t)brow * H_ + k0 + 32, 0, 1);
          __builtin_prefetch(WattnT + (size_t)brow * H_ + k0 + 32, 0, 1);
        }
        bf16x16 af, bw;
        af = frag_bf(xbf, arow, D_, k0, lane);
        bw = frag_bf(WxT, brow, H_, k0, lane);
        accx = __builtin_amdgcn_wmma_f32_16x16x32_bf16(false, af, false, bw,
                                                       (short)0, accx, false, false);
        af = frag_bf(hbf, arow, H_, k0, lane);
        bw = frag_bf(WhT, brow, H_, k0, lane);
        acch = __builtin_amdgcn_wmma_f32_16x16x32_bf16(false, af, false, bw,
                                                       (short)0, acch, false, false);
        af = frag_bf(attnbf, arow, H_, k0, lane);
        bw = frag_bf(WattnT, brow, H_, k0, lane);
        acca = __builtin_amdgcn_wmma_f32_16x16x32_bf16(false, af, false, bw,
                                                       (short)0, acca, false, false);
      }
      // C/D layout: VGPR j -> row j (lanes 0-15) / row j+8 (lanes 16-31), col = lane&15
      float* aoptr = a_ws + (size_t)(rbase + srow) * FH + cbase + scol;
#pragma unroll
      for (int j = 0; j < 8; ++j)
        aoptr[(size_t)j * FH] = accx[j] + acch[j] + acca[j];
    }
    grid_sync(cnt, gen);

    // -------- phase C: LSTM gates, update h/c, write out[n][t][:] --------
    {
#pragma unroll
      for (int e = 0; e < 2; ++e) {
        int idx = blockIdx.x * 512 + e * 256 + tid;  // 0..65535 over (n, j)
        int nn = idx >> 9;
        int j  = idx & 511;
        const float* ar = a_ws + (size_t)nn * FH;
        float ai  = ar[j]        + b[j];
        float afv = ar[512 + j]  + b[512 + j];
        float aov = ar[1024 + j] + b[1024 + j];
        float agv = ar[1536 + j] + b[1536 + j];
        float ig = 1.f / (1.f + __expf(-ai));
        float fg = 1.f / (1.f + __expf(-afv));
        float og = 1.f / (1.f + __expf(-aov));
        float gg = tanhf(agv);
        float cn = fg * c_ws[idx] + ig * gg;
        float hn = og * tanhf(cn);
        c_ws[idx] = cn;
        h_ws[idx] = hn;
        out[(size_t)nn * T_ * H_ + (size_t)t * H_ + j] = hn;
      }
    }
    grid_sync(cnt, gen);
  }
}

extern "C" void kernel_launch(void* const* d_in, const int* in_sizes, int n_in,
                              void* d_out, int out_size, void* d_ws, size_t ws_size,
                              hipStream_t stream) {
  (void)in_sizes; (void)n_in; (void)out_size; (void)ws_size;
  const float* x     = (const float*)d_in[0];
  const float* A     = (const float*)d_in[1];
  const float* Wx    = (const float*)d_in[2];
  const float* Wh    = (const float*)d_in[3];
  const float* Wattn = (const float*)d_in[4];
  const float* b     = (const float*)d_in[5];
  float* out = (float*)d_out;

  // workspace layout (~8.3 MB total)
  char* w = (char*)d_ws;
  unsigned* bar = (unsigned*)w;                           // barrier state
  float* h_ws = (float*)(w + 256);                        // N*H f32
  float* c_ws = h_ws + N_ * H_;                           // N*H f32
  float* a_ws = c_ws + N_ * H_;                           // N*4H f32
  unsigned short* hbf    = (unsigned short*)(a_ws + (size_t)N_ * FH);  // N*H bf16
  unsigned short* attnbf = hbf + N_ * H_;                 // N*H bf16
  unsigned short* xbf    = attnbf + N_ * H_;              // N*D bf16 (current timestep)
  unsigned short* WxT    = xbf + N_ * D_;                 // 4H*K bf16 (transposed)
  unsigned short* WhT    = WxT + (size_t)FH * H_;
  unsigned short* WattnT = WhT + (size_t)FH * H_;

  hipMemsetAsync(bar, 0, 256, stream);
  hipLaunchKernelGGL(attn_lstm_persistent, dim3(NB), dim3(TPB), 0, stream,
                     x, A, Wx, Wh, Wattn, b, out,
                     bar, h_ws, c_ws, a_ws, hbf, attnbf, xbf, WxT, WhT, WattnT);
}
